// RGTGraphRefiner_74663711474353
// MI455X (gfx1250) — compile-verified, matrix operand order checked by reading, not run on hardware
//
#include <hip/hip_runtime.h>
#include <hip/hip_bf16.h>
#include <hip/hip_fp16.h>

typedef __attribute__((ext_vector_type(16))) _Float16 v16h;
typedef __attribute__((ext_vector_type(8)))  _Float16 v8h;
typedef __attribute__((ext_vector_type(8)))  float    v8f;

#define NNODE   262144      // B*H*W
#define NPER    65536       // H*W
#define HGT     256
#define WID     256
#define CCH     64
#define HIDDEN  128
#define OUTC    64
#define KIN     192
#define INV2S2  78.1249999993896f   /* 1/(2*0.08^2 + 1e-12) */

// ---------------------------------------------------------------- utilities

__device__ __forceinline__ float raw_edge_w(int n, int e, const float* __restrict__ r01, int* vout) {
    int pix = n & (NPER - 1);
    int i = pix >> 8, j = pix & 255;
    int v = n, valid = 0;
    if      (e == 0) { if (j > 0)   { v = n - 1;   valid = 1; } }
    else if (e == 1) { if (j < 255) { v = n + 1;   valid = 1; } }
    else if (e == 2) { if (i > 0)   { v = n - 256; valid = 1; } }
    else             { if (i < 255) { v = n + 256; valid = 1; } }
    *vout = v;
    if (!valid) return 0.0f;
    float d = r01[n] - r01[v];
    return expf(-d * d * INV2S2);
}

// ------------------------------------------------------- 1. per-image minmax
__global__ void k_minmax(const float* __restrict__ rgt, float* __restrict__ scal) {
    int b = blockIdx.x;
    const float* p = rgt + (size_t)b * NPER;
    float mn = 3.4e38f, mx = -3.4e38f;
    for (int i = threadIdx.x; i < NPER; i += blockDim.x) {
        float v = p[i];
        mn = fminf(mn, v);
        mx = fmaxf(mx, v);
    }
    __shared__ float smn[256], smx[256];
    smn[threadIdx.x] = mn; smx[threadIdx.x] = mx;
    __syncthreads();
    for (int s = 128; s > 0; s >>= 1) {
        if (threadIdx.x < s) {
            smn[threadIdx.x] = fminf(smn[threadIdx.x], smn[threadIdx.x + s]);
            smx[threadIdx.x] = fmaxf(smx[threadIdx.x], smx[threadIdx.x + s]);
        }
        __syncthreads();
    }
    if (threadIdx.x == 0) { scal[b] = smn[0]; scal[4 + b] = smx[0]; }
}

// ------------------------------------------------------- 2. normalized rgt
__global__ void k_r01(const float* __restrict__ rgt, const float* __restrict__ scal,
                      float* __restrict__ r01) {
    int n = blockIdx.x * blockDim.x + threadIdx.x;
    if (n >= NNODE) return;
    int b = n >> 16;
    float mn = scal[b], mx = scal[4 + b];
    r01[n] = (rgt[n] - mn) / fmaxf(mx - mn, 1e-12f);
}

// ------------------------------------------------------- 3. degrees
__global__ void k_deg(const float* __restrict__ r01, float* __restrict__ dinv,
                      float* __restrict__ dinvsl) {
    int n = blockIdx.x * blockDim.x + threadIdx.x;
    if (n >= NNODE) return;
    float deg = 0.0f;
    for (int e = 0; e < 4; ++e) { int v; deg += raw_edge_w(n, e, r01, &v); }
    dinv[n]   = rsqrtf(deg + 1e-12f);   // diffusion norm (no self loop)
    dinvsl[n] = rsqrtf(deg + 1.0f);     // GCN norm (self loop weight 1)
}

// ------------------------------------------------------- 4. NCHW -> node-major (+f16 copy)
__global__ void k_transpose(const float* __restrict__ feat, float* __restrict__ X0,
                            _Float16* __restrict__ Xh) {
    __shared__ float tile[64][65];
    int b    = blockIdx.x >> 10;            // 1024 pixel tiles per image
    int pix0 = (blockIdx.x & 1023) << 6;    // 64 pixels per tile
    int t = threadIdx.x;
    #pragma unroll
    for (int it = 0; it < 16; ++it) {
        int idx = it * 256 + t;
        int c = idx >> 6, p = idx & 63;
        tile[c][p] = feat[(((size_t)(b * 64 + c)) << 16) + pix0 + p];
    }
    __syncthreads();
    #pragma unroll
    for (int it = 0; it < 16; ++it) {
        int idx = it * 256 + t;
        int p = idx >> 6, c = idx & 63;
        size_t n = (size_t)b * NPER + pix0 + p;
        float v = tile[c][p];
        X0[n * 64 + c] = v;
        Xh[n * 192 + c] = (_Float16)v;
    }
}

// ------------------------------------------------------- 5. propagate (diffusion / GCN)
__global__ void k_prop(const float* __restrict__ hsrc, const float* __restrict__ r01,
                       const float* __restrict__ dsc, const float* __restrict__ bias,
                       float* __restrict__ outf32, _Float16* __restrict__ outf16,
                       int Cc, int f16stride, int withself) {
    int npb  = blockDim.x / Cc;
    int slot = threadIdx.x / Cc;
    int c    = threadIdx.x % Cc;
    int n    = blockIdx.x * npb + slot;
    __shared__ float scoef[4][8];
    __shared__ int   snbr[4][4];
    if (c < 4) {
        int v;
        float w = raw_edge_w(n, c, r01, &v);
        snbr[slot][c]  = v;
        scoef[slot][c] = w * dsc[v] * dsc[n];
    } else if (c == 4) {
        float du = dsc[n];
        scoef[slot][4] = withself ? du * du : 0.0f;
    }
    __syncthreads();
    float acc = bias ? bias[c] : 0.0f;
    acc += scoef[slot][4] * hsrc[(size_t)n * Cc + c];
    #pragma unroll
    for (int e = 0; e < 4; ++e)
        acc += scoef[slot][e] * hsrc[(size_t)snbr[slot][e] * Cc + c];
    if (outf32) outf32[(size_t)n * Cc + c] = acc;
    if (outf16) outf16[(size_t)n * f16stride + c] = (_Float16)acc;
}

// ------------------------------------------------------- 6. weight prep (transpose + f16)
__global__ void k_prepw(const float* __restrict__ W1, const float* __restrict__ W2,
                        _Float16* __restrict__ W1T, _Float16* __restrict__ W2T) {
    for (int idx = threadIdx.x; idx < HIDDEN * KIN; idx += blockDim.x) {
        int nn = idx / KIN, k = idx % KIN;
        W1T[idx] = (_Float16)W1[k * HIDDEN + nn];
    }
    for (int idx = threadIdx.x; idx < OUTC * HIDDEN; idx += blockDim.x) {
        int nn = idx / HIDDEN, k = idx % HIDDEN;
        W2T[idx] = (_Float16)W2[k * OUTC + nn];
    }
}

// ------------------------------------------------------- 7. WMMA GEMM (C = A * B^T_stored)
// A: (rows x K) f16 row-major; BT: (ncols x K) f16 row-major (i.e. B column-major)
// one wave per 16x16 output tile; blockDim.x = (ncols/16)*32
__global__ void k_gemm_wmma(const _Float16* __restrict__ A, const _Float16* __restrict__ BT,
                            float* __restrict__ C, int K, int ncols) {
    int wave = threadIdx.x >> 5;
    int lane = threadIdx.x & 31;
    int row0 = blockIdx.x << 4;
    int col0 = wave << 4;
    int mr = lane & 15;                 // A row (lanes 0-15) / B col (same index role)
    int kb = (lane >> 4) << 3;          // K sub-base: 0 or 8
    const _Float16* ap = A  + (size_t)(row0 + mr) * K + kb;
    const _Float16* bp = BT + (size_t)(col0 + mr) * K + kb;
    v8f acc = {};
    for (int kk = 0; kk < K; kk += 32) {
        __builtin_prefetch(ap + kk + 32, 0, 1);
        __builtin_prefetch(bp + kk + 32, 0, 1);
        v8h alo = *(const v8h*)(ap + kk);
        v8h ahi = *(const v8h*)(ap + kk + 16);
        v8h blo = *(const v8h*)(bp + kk);
        v8h bhi = *(const v8h*)(bp + kk + 16);
        v16h a = __builtin_shufflevector(alo, ahi, 0,1,2,3,4,5,6,7,8,9,10,11,12,13,14,15);
        v16h b = __builtin_shufflevector(blo, bhi, 0,1,2,3,4,5,6,7,8,9,10,11,12,13,14,15);
        acc = __builtin_amdgcn_wmma_f32_16x16x32_f16(false, a, false, b,
                                                     (short)0, acc, false, false);
    }
    int nidx = lane & 15;
    int mrow = (lane >> 4) << 3;
    float* cp = C + (size_t)(row0 + mrow) * ncols + col0 + nidx;
    #pragma unroll
    for (int r = 0; r < 8; ++r) cp[(size_t)r * ncols] = acc[r];
}

// ------------------------------------------------------- 8. LayerNorm + ReLU -> f16
__global__ void k_ln(const float* __restrict__ y, const float* __restrict__ gamma,
                     const float* __restrict__ beta, _Float16* __restrict__ yh) {
    int slot = threadIdx.x >> 7;        // 2 nodes per 256-thread block
    int c    = threadIdx.x & 127;
    int n    = (blockIdx.x << 1) + slot;
    float v = y[(size_t)n * HIDDEN + c];
    __shared__ float s1[2][128], s2[2][128];
    s1[slot][c] = v; s2[slot][c] = v * v;
    __syncthreads();
    for (int s = 64; s > 0; s >>= 1) {
        if (c < s) {
            s1[slot][c] += s1[slot][c + s];
            s2[slot][c] += s2[slot][c + s];
        }
        __syncthreads();
    }
    float mu  = s1[slot][0] * (1.0f / 128.0f);
    float var = s2[slot][0] * (1.0f / 128.0f) - mu * mu;
    float t = (v - mu) * rsqrtf(var + 1e-5f) * gamma[c] + beta[c];
    yh[(size_t)n * HIDDEN + c] = (_Float16)fmaxf(t, 0.0f);
}

// ------------------------------------------------------- 9. final propagate + bias + residual + NCHW
__global__ void k_prop2res(const float* __restrict__ h2, const float* __restrict__ r01,
                           const float* __restrict__ dscsl, const float* __restrict__ b2,
                           const float* __restrict__ feat, const float* __restrict__ gatep,
                           float* __restrict__ out) {
    int slot = threadIdx.x >> 6;        // 4 nodes per block (C=64)
    int c    = threadIdx.x & 63;
    int n    = (blockIdx.x << 2) + slot;
    __shared__ float scoef[4][8];
    __shared__ int   snbr[4][4];
    if (c < 4) {
        int v;
        float w = raw_edge_w(n, c, r01, &v);
        snbr[slot][c]  = v;
        scoef[slot][c] = w * dscsl[v] * dscsl[n];
    } else if (c == 4) {
        float du = dscsl[n];
        scoef[slot][4] = du * du;
    }
    __syncthreads();
    float acc = b2[c] + scoef[slot][4] * h2[(size_t)n * OUTC + c];
    #pragma unroll
    for (int e = 0; e < 4; ++e)
        acc += scoef[slot][e] * h2[(size_t)snbr[slot][e] * OUTC + c];
    int b = n >> 16, pix = n & (NPER - 1);
    size_t fidx = (((size_t)(b * 64 + c)) << 16) + pix;
    float f = feat[fidx];
    float g = fminf(fmaxf(gatep[0], 0.0f), 1.0f);
    out[fidx] = f + g * (acc - f);
}

// ---------------------------------------------------------------- launch

extern "C" void kernel_launch(void* const* d_in, const int* in_sizes, int n_in,
                              void* d_out, int out_size, void* d_ws, size_t ws_size,
                              hipStream_t stream) {
    const float* feat  = (const float*)d_in[0];
    const float* rgt   = (const float*)d_in[1];
    const float* W1    = (const float*)d_in[2];
    const float* b1    = (const float*)d_in[3];
    const float* gamma = (const float*)d_in[4];
    const float* beta  = (const float*)d_in[5];
    const float* W2    = (const float*)d_in[6];
    const float* b2    = (const float*)d_in[7];
    const float* gate  = (const float*)d_in[8];
    float* out = (float*)d_out;

    char* ws = (char*)d_ws;
    const size_t NB = (size_t)NNODE * 4;
    const size_t oScal = 0;
    const size_t oR01  = 4096;
    const size_t oDinv = oR01  + NB;
    const size_t oDsl  = oDinv + NB;
    const size_t oR1   = oDsl  + NB;                         // 128 MB: X0,X1 -> H1
    const size_t oR2   = oR1 + (size_t)NNODE * HIDDEN * 4;   //  96 MB: Xh -> Y1h
    const size_t oR3   = oR2 + (size_t)NNODE * KIN * 2;      // 128 MB: Y1 -> H2
    const size_t oW1T  = oR3 + (size_t)NNODE * HIDDEN * 4;
    const size_t oW2T  = oW1T + (size_t)HIDDEN * KIN * 2;

    float*    scal   = (float*)(ws + oScal);
    float*    r01    = (float*)(ws + oR01);
    float*    dinv   = (float*)(ws + oDinv);
    float*    dinvsl = (float*)(ws + oDsl);
    float*    X0     = (float*)(ws + oR1);
    float*    X1     = X0 + (size_t)NNODE * 64;
    _Float16* Xh     = (_Float16*)(ws + oR2);
    float*    H1     = (float*)(ws + oR1);                   // alias X0/X1
    float*    Y1     = (float*)(ws + oR3);
    _Float16* Y1h    = (_Float16*)(ws + oR2);                // alias Xh
    float*    H2     = (float*)(ws + oR3);                   // alias Y1
    _Float16* W1T    = (_Float16*)(ws + oW1T);
    _Float16* W2T    = (_Float16*)(ws + oW2T);

    // 1-3: edge-weight scalars
    k_minmax<<<4, 256, 0, stream>>>(rgt, scal);
    k_r01<<<NNODE / 256, 256, 0, stream>>>(rgt, scal, r01);
    k_deg<<<NNODE / 256, 256, 0, stream>>>(r01, dinv, dinvsl);
    // 4: node-major features (f32 for diffusion, f16 col-block 0 of Xcat)
    k_transpose<<<4096, 256, 0, stream>>>(feat, X0, Xh);
    // 5: two diffusion steps -> Xcat col-blocks 1,2 (f16)
    k_prop<<<(NNODE * 64) / 256, 256, 0, stream>>>(X0, r01, dinv, nullptr,
                                                   X1, Xh + 64, 64, 192, 0);
    k_prop<<<(NNODE * 64) / 256, 256, 0, stream>>>(X1, r01, dinv, nullptr,
                                                   nullptr, Xh + 128, 64, 192, 0);
    // 6: weights -> f16 transposed
    k_prepw<<<1, 256, 0, stream>>>(W1, W2, W1T, W2T);
    // 7: GEMM1 (N x 192) @ (192 x 128)   [WMMA f16 -> f32 acc]
    k_gemm_wmma<<<NNODE / 16, 256, 0, stream>>>(Xh, W1T, H1, KIN, HIDDEN);
    // 8: GCN propagate 1 (+b1)
    k_prop<<<(NNODE * 128) / 256, 256, 0, stream>>>(H1, r01, dinvsl, b1,
                                                    Y1, nullptr, 128, 0, 1);
    // 9: LayerNorm + ReLU -> f16
    k_ln<<<NNODE / 2, 256, 0, stream>>>(Y1, gamma, beta, Y1h);
    // 10: GEMM2 (N x 128) @ (128 x 64)
    k_gemm_wmma<<<NNODE / 16, 128, 0, stream>>>(Y1h, W2T, H2, HIDDEN, OUTC);
    // 11: GCN propagate 2 (+b2) fused with gated residual + NCHW store
    k_prop2res<<<NNODE / 4, 256, 0, stream>>>(H2, r01, dinvsl, b2, feat, gate, out);
}
